// HopfieldLayer_37417755083722
// MI455X (gfx1250) — compile-verified, matrix operand order checked by reading, not run on hardware
//
#include <hip/hip_runtime.h>
#include <math.h>

// Problem dimensions (fixed by the reference)
#define DD    2048
#define BATCH 512
#define NEL   (BATCH * DD)   // 1,048,576 elements in x
#define WEL   (DD * DD)      // 4,194,304 elements in W

typedef __attribute__((ext_vector_type(16))) __bf16 v16bf;
typedef __attribute__((ext_vector_type(8)))  __bf16 v8bf;
typedef __attribute__((ext_vector_type(8)))  float  v8f;
typedef __attribute__((ext_vector_type(4)))  int    v4i;

typedef __attribute__((address_space(1))) v4i* gv4i_p;   // global 16B vector ptr
typedef __attribute__((address_space(3))) v4i* lv4i_p;   // LDS 16B vector ptr

struct SolverState {
  float    dt;
  int      done;
  unsigned err_bits;   // bit pattern of max|x5-x4| (>=0 so uint-monotone)
  unsigned spd_bits;   // bit pattern of max|x5-x|
  int      accept;
};

// ---------- CDNA5 async-to-LDS path (probe via __has_builtin) ---------------
#if defined(__has_builtin)
#if __has_builtin(__builtin_amdgcn_global_load_async_to_lds_b128)
#define HAVE_ASYNC_LDS 1
#else
#define HAVE_ASYNC_LDS 0
#endif
#else
#define HAVE_ASYNC_LDS 0
#endif

#if HAVE_ASYNC_LDS
#pragma message("gfx1250: using global_load_async_to_lds_b128 builtins")
#else
#pragma message("gfx1250: async-to-LDS builtin unavailable; synchronous LDS staging")
#endif

#if defined(__has_builtin) && __has_builtin(__builtin_amdgcn_s_wait_asynccnt)
#define WAIT_ASYNC(n) __builtin_amdgcn_s_wait_asynccnt(n)
#elif HAVE_ASYNC_LDS
#define WAIT_ASYNC(n) asm volatile("s_wait_asynccnt %0" ::"i"(n))
#else
#define WAIT_ASYNC(n) ((void)0)
#endif

static __device__ __forceinline__ void async_copy16(const __bf16* gsrc, __bf16* ldst) {
#if HAVE_ASYNC_LDS
  __builtin_amdgcn_global_load_async_to_lds_b128(
      (gv4i_p)(void*)gsrc, (lv4i_p)(void*)ldst, 0, 0);
#else
  *(v8bf*)ldst = *(const v8bf*)gsrc;
#endif
}

// ---------- bf16 helpers (manual RNE split; no reliance on __bf16 casts) ----
static __device__ __forceinline__ __bf16 f2bf(float f) {
  unsigned u = __float_as_uint(f);
  unsigned r = (u + 0x7FFFu + ((u >> 16) & 1u)) >> 16;  // round-nearest-even
  unsigned short hv = (unsigned short)r;
  __bf16 out;
  __builtin_memcpy(&out, &hv, 2);
  return out;
}
static __device__ __forceinline__ float bf2f(__bf16 h) {
  unsigned short hv;
  __builtin_memcpy(&hv, &h, 2);
  return __uint_as_float(((unsigned)hv) << 16);
}
static __device__ __forceinline__ v16bf cat8(v8bf a, v8bf b) {
  return __builtin_shufflevector(a, b, 0,1,2,3,4,5,6,7,8,9,10,11,12,13,14,15);
}

// ---------- init kernels ----------------------------------------------------
__global__ void k_split_bf16(const float* __restrict__ src,
                             __bf16* __restrict__ hi, __bf16* __restrict__ lo,
                             int n) {
  int i = blockIdx.x * blockDim.x + threadIdx.x;
  if (i >= n) return;
  float v = src[i];
  __bf16 h = f2bf(v);
  hi[i] = h;
  lo[i] = f2bf(v - bf2f(h));
}

__global__ void k_copy_f32(const float* __restrict__ src, float* __restrict__ dst, int n) {
  int i = blockIdx.x * blockDim.x + threadIdx.x;
  if (i < n) dst[i] = src[i];
}

__global__ void k_init_state(SolverState* st) {
  st->dt = 0.1f; st->done = 0; st->err_bits = 0u; st->spd_bits = 0u; st->accept = 0;
}

__global__ void k_prestep(SolverState* st) {
  st->err_bits = 0u; st->spd_bits = 0u;
}

// ---------- stage preparation: x_stage = x + dt*sum(c_j k_j); reflect -------
__global__ void k_stage_prep(const float* __restrict__ x,
                             const float* __restrict__ kbase,  // k1..k5 contiguous, NEL each
                             int nk, float c0, float c1, float c2, float c3, float c4,
                             const SolverState* __restrict__ st,
                             __bf16* __restrict__ yh, __bf16* __restrict__ yl,
                             float* __restrict__ s) {
  int i = blockIdx.x * blockDim.x + threadIdx.x;
  if (i >= NEL) return;
  const float dt = st->dt;
  const float cs[5] = {c0, c1, c2, c3, c4};
  float acc = 0.f;
#pragma unroll
  for (int j = 0; j < 5; ++j)
    if (j < nk) acc += cs[j] * kbase[(size_t)j * NEL + i];
  float xs = x[i] + dt * acc;
  int cell = (int)xs;                       // truncation toward zero == astype(int32)
  bool even = ((cell & 1) == 0);            // two's-complement parity matches jnp i%2==0
  float y  = even ? xs   : (1.0f - xs);
  float sg = even ? 1.0f : -1.0f;
  __bf16 h = f2bf(y);
  yh[i] = h;
  yl[i] = f2bf(y - bf2f(h));
  s[i]  = sg;
}

// ---------- k_j = s .* (Y @ W + b), split-bf16 WMMA GEMM --------------------
// Block: 8 waves = 128(M) x 64(N) tile. B tile (64 W-rows x 32 k, hi+lo) is
// staged in LDS, double-buffered via async-to-LDS, shared by all 8 waves.
// W symmetric => B[k][n] == W[n][k]: contiguous row reads, no transpose.
#define BRS 40   // LDS row stride in bf16 (80B): 20*r mod 64 is a permutation
__global__ __launch_bounds__(256) void k_pvf_gemm(
    const __bf16* __restrict__ Yh, const __bf16* __restrict__ Yl,
    const __bf16* __restrict__ Wh, const __bf16* __restrict__ Wl,
    const float*  __restrict__ S,  const float* __restrict__ bias,
    float* __restrict__ K) {
  __shared__ __bf16 Bh[2][64 * BRS];
  __shared__ __bf16 Bl[2][64 * BRS];

  const int tid  = threadIdx.x;
  const int lane = tid & 31;
  const int wave = tid >> 5;
  const int bm   = blockIdx.x & 3;   // 4 M-blocks of 128 rows
  const int bn   = blockIdx.x >> 2;  // 32 N-tiles of 64 cols
  const int r    = lane & 15;
  const int half = lane >> 4;        // K half-pattern of this lane (ISA layout)
  const int col0 = bn << 6;

  // staging assignment: thread -> (row 0..63, 16B chunk 0..3)
  const int srow   = tid >> 2;
  const int schunk = tid & 3;
  const __bf16* gh = Wh + (size_t)(col0 + srow) * DD + (schunk << 3);
  const __bf16* gl = Wl + (size_t)(col0 + srow) * DD + (schunk << 3);
  __bf16* lh = &Bh[0][srow * BRS + (schunk << 3)];
  __bf16* ll = &Bl[0][srow * BRS + (schunk << 3)];

  const int mbase = (bm << 7) + (wave << 4);
  const __bf16* yh_row = Yh + (size_t)(mbase + r) * DD;
  const __bf16* yl_row = Yl + (size_t)(mbase + r) * DD;

  v8f c[4] = {};
  const int NT = DD / 32;  // 64 k-tiles

  // prologue: stage k-tile 0 into buffer 0
  async_copy16(gh, lh);
  async_copy16(gl, ll);

  for (int kt = 0; kt < NT; ++kt) {
    const int cur = kt & 1;
    if (kt + 1 < NT) {       // stage next tile into the other buffer
      const int nb = (kt + 1) & 1;
      async_copy16(gh + ((kt + 1) << 5), lh + nb * (64 * BRS));
      async_copy16(gl + ((kt + 1) << 5), ll + nb * (64 * BRS));
      WAIT_ASYNC(2);         // current tile's 2 async loads retired
    } else {
      WAIT_ASYNC(0);
    }
    __syncthreads();         // current buffer visible to all waves

    const int o0 = (kt << 5) + (half << 3);
    const int o1 = o0 + 16;
    if (kt + 1 < NT) {
      __builtin_prefetch(yh_row + o0 + 32, 0, 3);
      __builtin_prefetch(yl_row + o0 + 32, 0, 3);
    }
    v16bf ah = cat8(*(const v8bf*)(yh_row + o0), *(const v8bf*)(yh_row + o1));
    v16bf al = cat8(*(const v8bf*)(yl_row + o0), *(const v8bf*)(yl_row + o1));
    const __bf16* bhbase = &Bh[cur][0];
    const __bf16* blbase = &Bl[cur][0];
#pragma unroll
    for (int n = 0; n < 4; ++n) {
      const int boff = ((n << 4) + r) * BRS + (half << 3);
      v16bf bh = cat8(*(const v8bf*)(bhbase + boff), *(const v8bf*)(bhbase + boff + 16));
      v16bf bl = cat8(*(const v8bf*)(blbase + boff), *(const v8bf*)(blbase + boff + 16));
      // split product: hi*hi + lo*hi + hi*lo  (~f32 accuracy)
      c[n] = __builtin_amdgcn_wmma_f32_16x16x32_bf16(false, ah, false, bh, (short)0, c[n], false, false);
      c[n] = __builtin_amdgcn_wmma_f32_16x16x32_bf16(false, al, false, bh, (short)0, c[n], false, false);
      c[n] = __builtin_amdgcn_wmma_f32_16x16x32_bf16(false, ah, false, bl, (short)0, c[n], false, false);
    }
    __syncthreads();         // protect current buffer before next overwrite
  }

  // C/D layout: VGPR e, lanes 0-15 -> M=e, lanes 16-31 -> M=e+8; N = lane&15
#pragma unroll
  for (int n = 0; n < 4; ++n) {
    int col = col0 + (n << 4) + r;
    float bv = bias[col];
#pragma unroll
    for (int e = 0; e < 8; ++e) {
      int gm = mbase + e + (half << 3);
      size_t idx = (size_t)gm * DD + col;
      K[idx] = S[idx] * (c[n][e] + bv);
    }
  }
}

// ---------- candidates + global max reductions ------------------------------
__global__ void k_step_candidates(const float* __restrict__ x,
                                  const float* __restrict__ kb,   // k1..k6
                                  float* __restrict__ x5out,
                                  SolverState* st) {
  const float B5[6] = {16.f/135.f, 0.f, 6656.f/12825.f, 28561.f/56430.f, -9.f/50.f, 2.f/55.f};
  const float B4[6] = {25.f/216.f, 0.f, 1408.f/2565.f,  2197.f/4104.f,  -0.2f,      0.f};
  int i = blockIdx.x * blockDim.x + threadIdx.x;
  float e = 0.f, sp = 0.f;
  if (i < NEL) {
    float dt = st->dt;
    float a5 = 0.f, a4 = 0.f;
#pragma unroll
    for (int j = 0; j < 6; ++j) {
      float kv = kb[(size_t)j * NEL + i];
      a5 += B5[j] * kv;
      a4 += B4[j] * kv;
    }
    float xi = x[i];
    float v5 = xi + dt * a5;
    float v4 = xi + dt * a4;
    x5out[i] = v5;
    e  = fabsf(v5 - v4);
    sp = fabsf(v5 - xi);
  }
#pragma unroll
  for (int off = 16; off > 0; off >>= 1) {
    e  = fmaxf(e,  __shfl_xor(e,  off, 32));
    sp = fmaxf(sp, __shfl_xor(sp, off, 32));
  }
  __shared__ float se[8], ss[8];
  int lane = threadIdx.x & 31, wv = threadIdx.x >> 5;
  if (lane == 0) { se[wv] = e; ss[wv] = sp; }
  __syncthreads();
  if (threadIdx.x == 0) {
    float me = se[0], ms = ss[0];
    for (int w = 1; w < (int)(blockDim.x >> 5); ++w) {
      me = fmaxf(me, se[w]);
      ms = fmaxf(ms, ss[w]);
    }
    atomicMax(&st->err_bits, __float_as_uint(me));
    atomicMax(&st->spd_bits, __float_as_uint(ms));
  }
}

// ---------- scalar controller (1 thread) ------------------------------------
__global__ void k_control(SolverState* st) {
  float err = __uint_as_float(st->err_bits);
  float spd = __uint_as_float(st->spd_bits);
  float dt  = st->dt;
  int done  = st->done;
  int accept = (err < 0.01f) ? 1 : 0;
  int step   = (accept && !done) ? 1 : 0;
  float speed = spd / dt;
  int done_new = (done || (step && (speed < 1e-3f))) ? 1 : 0;
  float scale = 0.9f * powf(0.01f / (err + 1e-12f), 0.2f);
  scale = fminf(fmaxf(scale, 0.1f), 4.0f);
  float dt_new = fmaxf(dt * scale, 0.1f);
  st->accept = step;
  st->done   = done_new;
  if (!done) st->dt = dt_new;   // dt frozen once done (uses OLD done, as in ref)
}

__global__ void k_commit(float* __restrict__ x, const float* __restrict__ x5,
                         const SolverState* __restrict__ st) {
  int i = blockIdx.x * blockDim.x + threadIdx.x;
  if (i >= NEL) return;
  if (st->accept) x[i] = x5[i];
}

// ---------- driver ----------------------------------------------------------
extern "C" void kernel_launch(void* const* d_in, const int* in_sizes, int n_in,
                              void* d_out, int out_size, void* d_ws, size_t ws_size,
                              hipStream_t stream) {
  const float* x0   = (const float*)d_in[0];
  const float* W    = (const float*)d_in[1];
  const float* bias = (const float*)d_in[2];
  float* X = (float*)d_out;                 // x lives in d_out (512x2048 f32)

  char* ws = (char*)d_ws;
  __bf16* Wh = (__bf16*)(ws);                                   // 8 MB
  __bf16* Wl = (__bf16*)(ws + (size_t)8 * 1024 * 1024);         // 8 MB
  __bf16* Yh = (__bf16*)(ws + (size_t)16 * 1024 * 1024);        // 2 MB
  __bf16* Yl = (__bf16*)(ws + (size_t)18 * 1024 * 1024);        // 2 MB
  float*  S  = (float*)(ws + (size_t)20 * 1024 * 1024);         // 4 MB
  float*  Kb = (float*)(ws + (size_t)24 * 1024 * 1024);         // 6 x 4 MB
  float*  X5 = (float*)(ws + (size_t)48 * 1024 * 1024);         // 4 MB
  SolverState* st = (SolverState*)(ws + (size_t)52 * 1024 * 1024);

  // RKF45 stage coefficients
  static const float A[6][5] = {
    {0.f, 0.f, 0.f, 0.f, 0.f},
    {0.25f, 0.f, 0.f, 0.f, 0.f},
    {3.f/32.f, 9.f/32.f, 0.f, 0.f, 0.f},
    {1932.f/2197.f, -7200.f/2197.f, 7296.f/2197.f, 0.f, 0.f},
    {439.f/216.f, -8.f, 3680.f/513.f, -845.f/4104.f, 0.f},
    {-8.f/27.f, 2.f, -3544.f/2565.f, 1859.f/4104.f, -11.f/40.f}
  };

  k_split_bf16<<<(WEL + 255) / 256, 256, 0, stream>>>(W, Wh, Wl, WEL);
  k_copy_f32<<<(NEL + 255) / 256, 256, 0, stream>>>(x0, X, NEL);
  k_init_state<<<1, 1, 0, stream>>>(st);

  for (int stepi = 0; stepi < 32; ++stepi) {
    k_prestep<<<1, 1, 0, stream>>>(st);
    for (int j = 0; j < 6; ++j) {
      k_stage_prep<<<NEL / 256, 256, 0, stream>>>(
          X, Kb, j, A[j][0], A[j][1], A[j][2], A[j][3], A[j][4], st, Yh, Yl, S);
      k_pvf_gemm<<<128, 256, 0, stream>>>(Yh, Yl, Wh, Wl, S, bias,
                                          Kb + (size_t)j * NEL);
    }
    k_step_candidates<<<NEL / 256, 256, 0, stream>>>(X, Kb, X5, st);
    k_control<<<1, 1, 0, stream>>>(st);
    k_commit<<<NEL / 256, 256, 0, stream>>>(X, X5, st);
  }
}
